// FrameStackMamba2_67267777790256
// MI455X (gfx1250) — matrix operand order repaired and checked by reading, not compiled, not run on hardware
//
#include <hip/hip_runtime.h>
#include <hip/hip_bf16.h>

#define D_STATE 64
#define HEADDIM 64
#define NHEADS 8
#define D_INNER 512
#define CONV_DIM 640
#define D_IN_PROJ 1160
#define D_IN_PROJ_PAD 1280     /* padded to multiple of 128 for the block tile */
#define D_MODEL 256
#define Bsz 16
#define Kseq 256
#define NTOK (Bsz * Kseq)      /* 4096 */
#define FRAME_DIM 188
#define FRAME_PAD 192
#define EPS 1e-5f

typedef __attribute__((ext_vector_type(16))) __bf16 v16bf;
typedef __attribute__((ext_vector_type(8)))  __bf16 v8bf;
typedef __attribute__((ext_vector_type(8)))  float  v8f;

__device__ __forceinline__ unsigned short f2bf(float f) {
    unsigned int u = __float_as_uint(f);
    u += 0x7FFFu + ((u >> 16) & 1u);            // round-to-nearest-even
    return (unsigned short)(u >> 16);
}
__device__ __forceinline__ float sigmoidf_(float x) { return 1.f / (1.f + __expf(-x)); }
__device__ __forceinline__ float siluf_(float x)    { return x * sigmoidf_(x); }

__device__ __forceinline__ v16bf load_afrag(const unsigned short* ap, int kk, int g) {
    v8bf lo = *reinterpret_cast<const v8bf*>(ap + kk + 8 * g);
    v8bf hi = *reinterpret_cast<const v8bf*>(ap + kk + 16 + 8 * g);
    return __builtin_shufflevector(lo, hi, 0, 1, 2, 3, 4, 5, 6, 7,
                                   8, 9, 10, 11, 12, 13, 14, 15);
}

// ---------------------------------------------------------------- pad+convert
__global__ void convert_pad_bf16(const float* __restrict__ src, unsigned short* __restrict__ dst,
                                 int Rs, int Rp, int C, int Cp) {
    int idx = blockIdx.x * blockDim.x + threadIdx.x;
    if (idx >= Rp * Cp) return;
    int r = idx / Cp, c = idx - r * Cp;
    float v = (r < Rs && c < C) ? src[(size_t)r * C + c] : 0.f;
    dst[idx] = f2bf(v);
}

// ---------------------------------------------------------------- embeddings
__global__ void embed_kernel(const float* __restrict__ fctx, const int* __restrict__ ictx,
                             const float* __restrict__ act_e, const float* __restrict__ jmp_e,
                             const float* __restrict__ chr_e, const float* __restrict__ stg_e,
                             const float* __restrict__ lc_e,  const float* __restrict__ hb_e,
                             const float* __restrict__ gnd_e, const float* __restrict__ la_e,
                             unsigned short* __restrict__ frame_bf) {
    int tok = blockIdx.x;            // 0..NTOK-1
    int j   = threadIdx.x;           // 0..FRAME_PAD-1
    const int* ic = ictx + (size_t)tok * 17;
    float v = 0.f;
    if      (j <  64) v = fctx[(size_t)tok * 64 + j];
    else if (j <  96) v = act_e[ic[0]  * 32 + (j - 64)];
    else if (j < 100) v = jmp_e[ic[1]  * 4  + (j - 96)];
    else if (j < 108) v = chr_e[ic[2]  * 8  + (j - 100)];
    else if (j < 110) v = lc_e [ic[3]  * 2  + (j - 108)];
    else if (j < 112) v = hb_e [ic[4]  * 2  + (j - 110)];
    else if (j < 114) v = gnd_e[ic[5]  * 2  + (j - 112)];
    else if (j < 122) v = la_e [ic[6]  * 8  + (j - 114)];
    else if (j < 154) v = act_e[ic[8]  * 32 + (j - 122)];
    else if (j < 158) v = jmp_e[ic[9]  * 4  + (j - 154)];
    else if (j < 166) v = chr_e[ic[10] * 8  + (j - 158)];
    else if (j < 168) v = lc_e [ic[11] * 2  + (j - 166)];
    else if (j < 170) v = hb_e [ic[12] * 2  + (j - 168)];
    else if (j < 172) v = gnd_e[ic[13] * 2  + (j - 170)];
    else if (j < 180) v = la_e [ic[14] * 8  + (j - 172)];
    else if (j < 188) v = stg_e[ic[16] * 8  + (j - 180)];
    frame_bf[(size_t)tok * FRAME_PAD + j] = f2bf(v);
}

// ---------------------------------------------------------------- WMMA GEMM
// C[m][n] = sum_k A[m][k] * W[n][k] (+ bias[n]) (+ Cin[m][n])
// A: M x Kp bf16 row-major, W: Np x Kp bf16 row-major (Np multiple of 128).
// Block = 8 waves covering a 128x128 output tile; each wave computes a 32x64
// tile = 2x4 WMMA tiles via v_wmma_f32_16x16x32_bf16 (8 wmma per k-step,
// ~1.5 b128 loads per wmma).
__global__ void gemm_bf16_wmma(const unsigned short* __restrict__ Abf,
                               const unsigned short* __restrict__ Wbf,
                               const float* __restrict__ bias,
                               const float* __restrict__ Cin,
                               float* __restrict__ Cout,
                               int M, int N, int Kp, int ldc) {
    int wave = threadIdx.x >> 5;
    int lane = threadIdx.x & 31;
    int g    = lane >> 4;            // half-wave group
    int lm   = lane & 15;
    int wm   = wave >> 1;            // 0..3
    int wn   = wave & 1;             // 0..1
    int m0   = blockIdx.y * 128 + wm * 32;   // wave rows   m0..m0+31
    int n0   = blockIdx.x * 128 + wn * 64;   // wave cols   n0..n0+63

    const unsigned short* ap0 = Abf + (size_t)(m0 + lm) * Kp;
    const unsigned short* ap1 = Abf + (size_t)(m0 + 16 + lm) * Kp;
    const unsigned short* bp[4];
#pragma unroll
    for (int t = 0; t < 4; ++t)
        bp[t] = Wbf + (size_t)(n0 + 16 * t + lm) * Kp;

    v8f acc[2][4] = {};
    for (int kk = 0; kk < Kp; kk += 32) {
        v16bf a0 = load_afrag(ap0, kk, g);
        v16bf a1 = load_afrag(ap1, kk, g);
        v16bf b[4];
#pragma unroll
        for (int t = 0; t < 4; ++t)
            b[t] = load_afrag(bp[t], kk, g);
#pragma unroll
        for (int t = 0; t < 4; ++t) {
            acc[0][t] = __builtin_amdgcn_wmma_f32_16x16x32_bf16(false, a0, false, b[t],
                                                                (short)0, acc[0][t], false, false);
            acc[1][t] = __builtin_amdgcn_wmma_f32_16x16x32_bf16(false, a1, false, b[t],
                                                                (short)0, acc[1][t], false, false);
        }
    }
#pragma unroll
    for (int i = 0; i < 2; ++i) {
#pragma unroll
        for (int t = 0; t < 4; ++t) {
            int n = n0 + 16 * t + lm;
            if (n >= N) continue;
            float bv = bias ? bias[n] : 0.f;
#pragma unroll
            for (int r = 0; r < 8; ++r) {
                int m = m0 + 16 * i + r + 8 * g;   // C layout: VGPR r -> M=r (+8 hi half)
                float v = acc[i][t][r] + bv;
                if (Cin) v += Cin[(size_t)m * ldc + n];
                Cout[(size_t)m * ldc + n] = v;
            }
        }
    }
}

// ---------------------------------------------------------------- rmsnorm -> bf16
__global__ void rmsnorm_bf16_kernel(const float* __restrict__ x, const float* __restrict__ w,
                                    unsigned short* __restrict__ out_bf) {
    __shared__ float red[256];
    int tok = blockIdx.x, d = threadIdx.x;
    float v = x[(size_t)tok * D_MODEL + d];
    red[d] = v * v;
    __syncthreads();
    for (int s = 128; s > 0; s >>= 1) { if (d < s) red[d] += red[d + s]; __syncthreads(); }
    float rs = rsqrtf(red[0] / (float)D_MODEL + EPS);
    out_bf[(size_t)tok * D_MODEL + d] = f2bf(v * rs * w[d]);
}

// ---------------------------------------------------------------- conv + silu
__global__ void conv_silu_kernel(const float* __restrict__ zx, const float* __restrict__ cw,
                                 const float* __restrict__ cb, float* __restrict__ xbc) {
    int idx = blockIdx.x * blockDim.x + threadIdx.x;
    if (idx >= NTOK * CONV_DIM) return;
    int c   = idx % CONV_DIM;
    int tok = idx / CONV_DIM;
    int k   = tok % Kseq;
    float v = cb[c];
#pragma unroll
    for (int j = 0; j < 4; ++j) {
        int kt = k - 3 + j;
        if (kt >= 0)
            v += zx[(size_t)(tok - 3 + j) * D_IN_PROJ + D_INNER + c] * cw[c * 4 + j];
    }
    xbc[idx] = siluf_(v);
}

// ---------------------------------------------------------------- dt = softplus
__global__ void dt_kernel(const float* __restrict__ zx, const float* __restrict__ dtb,
                          float* __restrict__ dt) {
    int idx = blockIdx.x * blockDim.x + threadIdx.x;
    if (idx >= NTOK * NHEADS) return;
    int h = idx % NHEADS, tok = idx / NHEADS;
    float x = zx[(size_t)tok * D_IN_PROJ + (D_IN_PROJ - NHEADS) + h] + dtb[h];
    dt[idx] = (x > 20.f) ? x : log1pf(__expf(x));
}

// ---------------------------------------------------------------- SSM scan
// one block per (b,h); 256 threads hold the 64x64 state (16 f32 each)
__global__ void scan_kernel(const float* __restrict__ xbc, const float* __restrict__ dt,
                            const float* __restrict__ alog, const float* __restrict__ dskip,
                            float* __restrict__ y) {
    int b = blockIdx.x >> 3, h = blockIdx.x & 7;
    int tid = threadIdx.x;
    int p = tid & 63, ng = tid >> 6;
    float Ah = -__expf(alog[h]);
    float Dh = dskip[h];
    __shared__ float sx[64], sB[64], sC[64], part[256];
    __shared__ float sdt;
    float st[16];
#pragma unroll
    for (int j = 0; j < 16; ++j) st[j] = 0.f;

    for (int t = 0; t < Kseq; ++t) {
        __syncthreads();
        size_t base = (size_t)(b * Kseq + t) * CONV_DIM;
        if      (tid <  64) sx[tid]       = xbc[base + h * HEADDIM + tid];
        else if (tid < 128) sB[tid - 64]  = xbc[base + D_INNER + (tid - 64)];
        else if (tid < 192) sC[tid - 128] = xbc[base + D_INNER + D_STATE + (tid - 128)];
        else if (tid == 192) sdt = dt[(size_t)(b * Kseq + t) * NHEADS + h];
        __syncthreads();
        float dtv = sdt;
        float dA  = __expf(Ah * dtv);
        float xp  = sx[p];
        float acc = 0.f;
#pragma unroll
        for (int j = 0; j < 16; ++j) {
            int n = ng * 16 + j;
            st[j] = st[j] * dA + dtv * sB[n] * xp;
            acc += st[j] * sC[n];
        }
        part[tid] = acc;
        __syncthreads();
        if (ng == 0) {
            float yv = part[p] + part[64 + p] + part[128 + p] + part[192 + p] + xp * Dh;
            y[(size_t)(b * Kseq + t) * D_INNER + h * HEADDIM + p] = yv;
        }
    }
}

// ---------------------------------------------------------------- gated rmsnorm -> bf16
__global__ void gated_norm_kernel(const float* __restrict__ y, const float* __restrict__ zx,
                                  const float* __restrict__ w, unsigned short* __restrict__ out_bf) {
    __shared__ float red[256];
    int tok = blockIdx.x, d = threadIdx.x;
    size_t yb = (size_t)tok * D_INNER;
    size_t zb = (size_t)tok * D_IN_PROJ;
    float t0 = y[yb + d]       * siluf_(zx[zb + d]);
    float t1 = y[yb + 256 + d] * siluf_(zx[zb + 256 + d]);
    red[d] = t0 * t0 + t1 * t1;
    __syncthreads();
    for (int s = 128; s > 0; s >>= 1) { if (d < s) red[d] += red[d + s]; __syncthreads(); }
    float rs = rsqrtf(red[0] / (float)D_INNER + EPS);
    out_bf[yb + d]       = f2bf(t0 * rs * w[d]);
    out_bf[yb + 256 + d] = f2bf(t1 * rs * w[256 + d]);
}

// ---------------------------------------------------------------- final norm + ctrl proj
__global__ void hvec_kernel(const float* __restrict__ x, const float* __restrict__ fw,
                            const float* __restrict__ nc, const float* __restrict__ cpw,
                            const float* __restrict__ cpb, float* __restrict__ hvec) {
    __shared__ float red[256];
    int b = blockIdx.x, d = threadIdx.x;
    float v = x[(size_t)(b * Kseq + Kseq - 1) * D_MODEL + d];
    red[d] = v * v;
    __syncthreads();
    for (int s = 128; s > 0; s >>= 1) { if (d < s) red[d] += red[d + s]; __syncthreads(); }
    float rs = rsqrtf(red[0] / (float)D_MODEL + EPS);
    float dot = 0.f;
#pragma unroll
    for (int j = 0; j < 16; ++j) dot += nc[b * 16 + j] * cpw[d * 16 + j];
    hvec[b * D_MODEL + d] = v * rs * fw[d] + dot + cpb[d];
}

// ---------------------------------------------------------------- output heads
__global__ void head_gemv(const float* __restrict__ hvec, const float* __restrict__ W,
                          const float* __restrict__ bias, float* __restrict__ out, int N) {
    int idx = blockIdx.x * blockDim.x + threadIdx.x;
    if (idx >= Bsz * N) return;
    int b = idx / N, n = idx - b * N;
    const float* hv = hvec + b * D_MODEL;
    const float* wr = W + (size_t)n * D_MODEL;
    float acc = bias[n];
    for (int j = 0; j < D_MODEL; ++j) acc += hv[j] * wr[j];
    out[idx] = acc;
}

// ================================================================ launch
extern "C" void kernel_launch(void* const* d_in, const int* in_sizes, int n_in,
                              void* d_out, int out_size, void* d_ws, size_t ws_size,
                              hipStream_t stream) {
    (void)in_sizes; (void)n_in; (void)out_size; (void)ws_size;
    const float* float_ctx    = (const float*)d_in[0];
    const float* next_ctrl    = (const float*)d_in[1];
    const float* action_e     = (const float*)d_in[2];
    const float* jumps_e      = (const float*)d_in[3];
    const float* char_e       = (const float*)d_in[4];
    const float* stage_e      = (const float*)d_in[5];
    const float* lc_e         = (const float*)d_in[6];
    const float* hb_e         = (const float*)d_in[7];
    const float* gnd_e        = (const float*)d_in[8];
    const float* la_e         = (const float*)d_in[9];
    const float* frame_proj_w = (const float*)d_in[10];
    const float* frame_proj_b = (const float*)d_in[11];
    const float* in_proj_w    = (const float*)d_in[12];
    const float* conv_w       = (const float*)d_in[13];
    const float* conv_b       = (const float*)d_in[14];
    const float* dt_bias      = (const float*)d_in[15];
    const float* A_log        = (const float*)d_in[16];
    const float* D_skip       = (const float*)d_in[17];
    const float* mnorm_w      = (const float*)d_in[18];
    const float* out_proj_w   = (const float*)d_in[19];
    const float* lnorm_w      = (const float*)d_in[20];
    const float* final_norm_w = (const float*)d_in[21];
    const float* ctrl_proj_w  = (const float*)d_in[22];
    const float* ctrl_proj_b  = (const float*)d_in[23];
    const float* cont_w       = (const float*)d_in[24];
    const float* cont_b       = (const float*)d_in[25];
    const float* bin_w        = (const float*)d_in[26];
    const float* bin_b        = (const float*)d_in[27];
    const float* p0a_w        = (const float*)d_in[28];
    const float* p0a_b        = (const float*)d_in[29];
    const float* p1a_w        = (const float*)d_in[30];
    const float* p1a_b        = (const float*)d_in[31];
    const float* p0j_w        = (const float*)d_in[32];
    const float* p0j_b        = (const float*)d_in[33];
    const float* p1j_w        = (const float*)d_in[34];
    const float* p1j_b        = (const float*)d_in[35];
    const int*   int_ctx      = (const int*)d_in[36];
    float* out = (float*)d_out;

    // ---- workspace carve (256B aligned chunks)
    char* wp = (char*)d_ws;
    auto alloc = [&](size_t bytes) -> void* {
        void* p = wp; wp += (bytes + 255) & ~(size_t)255; return p;
    };
    unsigned short* frame_bf = (unsigned short*)alloc((size_t)NTOK * FRAME_PAD * 2);
    unsigned short* fw_bf    = (unsigned short*)alloc((size_t)D_MODEL * FRAME_PAD * 2);
    unsigned short* ipw_bf   = (unsigned short*)alloc((size_t)2 * D_IN_PROJ_PAD * D_MODEL * 2);
    unsigned short* opw_bf   = (unsigned short*)alloc((size_t)2 * D_MODEL * D_INNER * 2);
    float*          x        = (float*)alloc((size_t)NTOK * D_MODEL * 4);
    unsigned short* xn_bf    = (unsigned short*)alloc((size_t)NTOK * D_MODEL * 2);
    float*          zx       = (float*)alloc((size_t)NTOK * D_IN_PROJ * 4);
    float*          xbc      = (float*)alloc((size_t)NTOK * CONV_DIM * 4);
    float*          dtbuf    = (float*)alloc((size_t)NTOK * NHEADS * 4);
    float*          ybuf     = (float*)alloc((size_t)NTOK * D_INNER * 4);
    unsigned short* y_bf     = (unsigned short*)alloc((size_t)NTOK * D_INNER * 2);
    float*          hvec     = (float*)alloc((size_t)Bsz * D_MODEL * 4);

    // ---- weight conversions (re-done every call: deterministic, no caching)
    {
        int tot = D_MODEL * FRAME_PAD;
        convert_pad_bf16<<<(tot + 255) / 256, 256, 0, stream>>>(
            frame_proj_w, fw_bf, D_MODEL, D_MODEL, FRAME_DIM, FRAME_PAD);
    }
    for (int l = 0; l < 2; ++l) {
        int tot = D_IN_PROJ_PAD * D_MODEL;
        convert_pad_bf16<<<(tot + 255) / 256, 256, 0, stream>>>(
            in_proj_w + (size_t)l * D_IN_PROJ * D_MODEL,
            ipw_bf + (size_t)l * D_IN_PROJ_PAD * D_MODEL,
            D_IN_PROJ, D_IN_PROJ_PAD, D_MODEL, D_MODEL);
        tot = D_MODEL * D_INNER;
        convert_pad_bf16<<<(tot + 255) / 256, 256, 0, stream>>>(
            out_proj_w + (size_t)l * D_MODEL * D_INNER,
            opw_bf + (size_t)l * D_MODEL * D_INNER,
            D_MODEL, D_MODEL, D_INNER, D_INNER);
    }

    // ---- frame encoding + projection
    embed_kernel<<<NTOK, FRAME_PAD, 0, stream>>>(float_ctx, int_ctx, action_e, jumps_e,
                                                 char_e, stage_e, lc_e, hb_e, gnd_e, la_e,
                                                 frame_bf);
    gemm_bf16_wmma<<<dim3(D_MODEL / 128, NTOK / 128), 256, 0, stream>>>(
        frame_bf, fw_bf, frame_proj_b, nullptr, x, NTOK, D_MODEL, FRAME_PAD, D_MODEL);

    // ---- Mamba layers
    for (int l = 0; l < 2; ++l) {
        rmsnorm_bf16_kernel<<<NTOK, 256, 0, stream>>>(x, lnorm_w + l * D_MODEL, xn_bf);
        gemm_bf16_wmma<<<dim3(D_IN_PROJ_PAD / 128, NTOK / 128), 256, 0, stream>>>(
            xn_bf, ipw_bf + (size_t)l * D_IN_PROJ_PAD * D_MODEL, nullptr, nullptr,
            zx, NTOK, D_IN_PROJ, D_MODEL, D_IN_PROJ);
        conv_silu_kernel<<<(NTOK * CONV_DIM + 255) / 256, 256, 0, stream>>>(
            zx, conv_w + l * CONV_DIM * 4, conv_b + l * CONV_DIM, xbc);
        dt_kernel<<<(NTOK * NHEADS + 255) / 256, 256, 0, stream>>>(
            zx, dt_bias + l * NHEADS, dtbuf);
        scan_kernel<<<Bsz * NHEADS, 256, 0, stream>>>(
            xbc, dtbuf, A_log + l * NHEADS, D_skip + l * NHEADS, ybuf);
        gated_norm_kernel<<<NTOK, 256, 0, stream>>>(ybuf, zx, mnorm_w + l * D_INNER, y_bf);
        gemm_bf16_wmma<<<dim3(D_MODEL / 128, NTOK / 128), 256, 0, stream>>>(
            y_bf, opw_bf + (size_t)l * D_MODEL * D_INNER, nullptr, x /*residual*/, x,
            NTOK, D_MODEL, D_INNER, D_MODEL);
    }

    // ---- final norm (last token only) + ctrl projection
    hvec_kernel<<<Bsz, 256, 0, stream>>>(x, final_norm_w, next_ctrl, ctrl_proj_w,
                                         ctrl_proj_b, hvec);

    // ---- heads (outputs concatenated in return order)
    head_gemv<<<(Bsz * 8   + 255) / 256, 256, 0, stream>>>(hvec, cont_w, cont_b, out + 0,     8);
    head_gemv<<<(Bsz * 6   + 255) / 256, 256, 0, stream>>>(hvec, bin_w,  bin_b,  out + 128,   6);
    head_gemv<<<(Bsz * 400 + 255) / 256, 256, 0, stream>>>(hvec, p0a_w,  p0a_b,  out + 224,   400);
    head_gemv<<<(Bsz * 400 + 255) / 256, 256, 0, stream>>>(hvec, p1a_w,  p1a_b,  out + 6624,  400);
    head_gemv<<<(Bsz * 8   + 255) / 256, 256, 0, stream>>>(hvec, p0j_w,  p0j_b,  out + 13024, 8);
    head_gemv<<<(Bsz * 8   + 255) / 256, 256, 0, stream>>>(hvec, p1j_w,  p1j_b,  out + 13152, 8);
}
